// DisentangledAtten_38817914421500
// MI455X (gfx1250) — matrix operand order repaired and checked by reading, not compiled, not run on hardware
//
#include <hip/hip_runtime.h>
#include <hip/hip_bf16.h>
#include <math.h>

typedef __bf16 bf16_t;
typedef __bf16 v16bf __attribute__((ext_vector_type(16)));
typedef __bf16 v8bf  __attribute__((ext_vector_type(8)));
typedef float  v8f   __attribute__((ext_vector_type(8)));
typedef unsigned int v4u __attribute__((ext_vector_type(4)));
typedef int v8i __attribute__((ext_vector_type(8)));
typedef int v4i __attribute__((ext_vector_type(4)));

#define WMMA_BF16(a, b, c) \
  __builtin_amdgcn_wmma_f32_16x16x32_bf16(false, (a), false, (b), (short)0, (c), false, false)

// ---------- WMMA operand loaders (wave32, 16x16x32 bf16) ----------
// A operand: lane l holds row M=l%16; element i -> K = (i>=8?16:0) + (l>=16?8:0) + (i&7)
__device__ __forceinline__ v16bf load_a16(const bf16_t* rowbase, int lane) {
  const int off = (lane >> 4) << 3;                // 0 or 8
  v8bf lo = *(const v8bf*)(rowbase + off);         // K = off + 0..7
  v8bf hi = *(const v8bf*)(rowbase + off + 16);    // K = off + 16..23
  return __builtin_shufflevector(lo, hi, 0,1,2,3,4,5,6,7,8,9,10,11,12,13,14,15);
}
// B operand: lane l holds column N=l%16; element j -> K = (j>=8?16:0) + (l>=16?8:0) + (j&7)  (rowbase = B^T row N)
__device__ __forceinline__ v16bf load_b16(const bf16_t* rowbase, int lane) {
  const int off = (lane >> 4) << 3;                // 0 or 8
  v8bf lo = *(const v8bf*)(rowbase + off);
  v8bf hi = *(const v8bf*)(rowbase + off + 16);
  return __builtin_shufflevector(lo, hi, 0,1,2,3,4,5,6,7,8,9,10,11,12,13,14,15);
}
// row reductions within 16-lane groups (one C-layout row per VGPR)
__device__ __forceinline__ float rowmax16(float v) {
  v = fmaxf(v, __shfl_xor(v, 1, 16));
  v = fmaxf(v, __shfl_xor(v, 2, 16));
  v = fmaxf(v, __shfl_xor(v, 4, 16));
  v = fmaxf(v, __shfl_xor(v, 8, 16));
  return v;
}
__device__ __forceinline__ float rowsum16(float v) {
  v += __shfl_xor(v, 1, 16);
  v += __shfl_xor(v, 2, 16);
  v += __shfl_xor(v, 4, 16);
  v += __shfl_xor(v, 8, 16);
  return v;
}

// ---------- TDM: DMA a 2D tile (bf16 elements) global -> LDS (ISA ch.8 D#) ----------
// tile_w elements contiguous per row, tile_h rows, row stride = stride_elems.
// Packs rows contiguously into LDS at lds_off ([tile_h][tile_w] bf16).
__device__ __forceinline__ void tdm_load_2d(const void* gptr, unsigned int lds_off,
                                            unsigned int tile_w, unsigned int tile_h,
                                            unsigned int stride_elems) {
  const unsigned long long ga = (unsigned long long)(uintptr_t)gptr;
  const unsigned int td = 1u << 20;                // tensor dims (accesses always in-bounds)
  v4u g0;
  g0[0] = 1u;                                      // count=1, user descriptor
  g0[1] = lds_off;                                 // lds_addr (bytes)
  g0[2] = (unsigned int)ga;                        // global_addr[31:0]
  g0[3] = (unsigned int)((ga >> 32) & 0x1ffffffu) | (2u << 30);  // global_addr[56:32], type=2
  v8i g1;
  g1[0] = 0x00010000;                              // workgroup_mask=0, data_size=1 (2 bytes)
  g1[1] = (int)((td & 0xffffu) << 16);             // tensor_dim0[15:0]
  g1[2] = (int)((td >> 16) | ((td & 0xffffu) << 16)); // tensor_dim0[31:16] | tensor_dim1[15:0]
  g1[3] = (int)((td >> 16) | (tile_w << 16));      // tensor_dim1[31:16] | tile_dim0
  g1[4] = (int)(tile_h & 0xffffu);                 // tile_dim1 (tile_dim2=0)
  g1[5] = (int)stride_elems;                       // tensor_dim0_stride[31:0]
  g1[6] = 0;                                       // stride hi / dim1_stride lo
  g1[7] = 0;
  v4i z4 = {0, 0, 0, 0};                           // groups 2/3 unused (2-D tile)
  v8i z8 = {0, 0, 0, 0, 0, 0, 0, 0};
  __builtin_amdgcn_tensor_load_to_lds(g0, g1, z4, z4, z8, 0);
}

// ---------- Kernel 1: fold W_c+W_p -> bf16 W^T[n][k]; sum biases ----------
__global__ __launch_bounds__(256) void da_prep_weights(
    const float* Wcq, const float* Wpq, const float* bcq, const float* bpq,
    const float* Wck, const float* Wpk, const float* bck, const float* bpk,
    const float* Wcv, const float* Wpv, const float* bcv, const float* bpv,
    bf16_t* wt, float* bsum) {
  int idx = blockIdx.x * 256 + threadIdx.x;            // 3*512*512 threads
  int which = idx >> 18;
  int rem = idx & 262143;
  int n = rem >> 9;
  int k = rem & 511;
  const float* Wc = (which == 0) ? Wcq : (which == 1) ? Wck : Wcv;
  const float* Wp = (which == 0) ? Wpq : (which == 1) ? Wpk : Wpv;
  wt[idx] = (bf16_t)(Wc[k * 512 + n] + Wp[k * 512 + n]);   // transpose: wt[which][n][k]
  if (k == 0) {
    const float* bc = (which == 0) ? bcq : (which == 1) ? bck : bcv;
    const float* bp = (which == 0) ? bpq : (which == 1) ? bpk : bpv;
    bsum[which * 512 + n] = bc[n] + bp[n];
  }
}

// ---------- Kernel 2: x -> bf16 ----------
__global__ __launch_bounds__(256) void da_convert_x(const float* x, bf16_t* xb) {
  int idx = blockIdx.x * 256 + threadIdx.x;            // 8*1024*512 threads
  xb[idx] = (bf16_t)x[idx];
}

// ---------- Kernel 3: QKV GEMMs. Block = 128(M) x 64(N); A staged via TDM ----------
__global__ __launch_bounds__(256) void da_qkv_gemm(
    const bf16_t* __restrict__ xb, const bf16_t* __restrict__ wt,
    const float* __restrict__ bsum,
    bf16_t* __restrict__ qb, bf16_t* __restrict__ kb, bf16_t* __restrict__ vtb) {
  __shared__ bf16_t atile[2][128 * 32];                // double-buffered A tile (8KB each)
  const int lane = threadIdx.x & 31;
  const int wid  = threadIdx.x >> 5;
  const int which = blockIdx.x >> 9;                   // 3 * 64 * 8 = 1536 blocks
  const int rem   = blockIdx.x & 511;
  const int mb  = rem >> 3;                            // 128-row tile (64 over 8192)
  const int n64 = rem & 7;                             // 64-col tile
  const bf16_t* W   = wt + (size_t)which * (512 * 512);
  const float* bias = bsum + which * 512;
  const int row0 = mb * 128 + wid * 16;
  const bf16_t* xtile = xb + (size_t)(mb * 128) * 512;
  const unsigned int lds[2] = {(unsigned int)(uintptr_t)&atile[0][0],
                               (unsigned int)(uintptr_t)&atile[1][0]};
  if (wid == 0) tdm_load_2d(xtile, lds[0], 32, 128, 512);

  v8f acc[4] = {v8f{}, v8f{}, v8f{}, v8f{}};
  for (int kk = 0; kk < 512; kk += 32) {
    const int cur = (kk >> 5) & 1;
    if (wid == 0) {
      if (kk + 32 < 512) {
        tdm_load_2d(xtile + kk + 32, lds[cur ^ 1], 32, 128, 512);
        __builtin_amdgcn_s_wait_tensorcnt(1);          // current tile done (TDM in-order)
      } else {
        __builtin_amdgcn_s_wait_tensorcnt(0);
      }
    }
    __syncthreads();
    const bf16_t* arow = &atile[cur][(wid * 16 + (lane & 15)) * 32];
    v16bf a = load_a16(arow, lane);
#pragma unroll
    for (int nt = 0; nt < 4; ++nt) {
      const int n = n64 * 64 + nt * 16 + (lane & 15);
      v16bf bm = load_b16(W + (size_t)n * 512 + kk, lane);
      acc[nt] = WMMA_BF16(a, bm, acc[nt]);
    }
    __syncthreads();                                   // all waves done before overwrite
  }
#pragma unroll
  for (int nt = 0; nt < 4; ++nt) {
    const int n = n64 * 64 + nt * 16 + (lane & 15);
    const float bv = bias[n];
#pragma unroll
    for (int r = 0; r < 8; ++r) {
      const int m = row0 + r + ((lane >> 4) << 3);     // global row = b*1024 + s
      const float val = acc[nt][r] + bv;
      if (which == 0) {
        qb[(size_t)m * 512 + n] = (bf16_t)val;
      } else if (which == 1) {
        kb[(size_t)m * 512 + n] = (bf16_t)val;
      } else {
        const int b = m >> 10, s = m & 1023;
        const int h = n64, dl = nt * 16 + (lane & 15);
        vtb[(((size_t)b * 8 + h) * 64 + dl) * 1024 + s] = (bf16_t)val;  // V^T [b][h][d][s]
      }
    }
  }
}

// ---------- Kernel 4: flash attention. Block = (b,h) x 128 queries; K/V staged via TDM ----------
__global__ __launch_bounds__(256) void da_attention(
    const bf16_t* __restrict__ qb, const bf16_t* __restrict__ kb,
    const bf16_t* __restrict__ vtb, float* __restrict__ out) {
  __shared__ bf16_t kt_lds[2][32 * 64];                // K tile  [key][d]   (4KB each)
  __shared__ bf16_t vt_lds[2][64 * 32];                // V^T tile [d][key]  (4KB each)
  __shared__ bf16_t plds[8][16 * 32];                  // per-wave P staging (C->A relayout)
  const int lane = threadIdx.x & 31;
  const int wid  = threadIdx.x >> 5;
  const int bh = blockIdx.x >> 3;                      // 512 blocks = 64 (b,h) x 8 qgroups
  const int b  = bh >> 3;
  const int h  = bh & 7;
  const int qg = blockIdx.x & 7;
  const int qt = qg * 8 + wid;                         // this wave's 16-query tile
  const float scale = 0.04419417382415922f;            // 1/sqrt(512)

  const bf16_t* qrow = qb + (size_t)(b * 1024 + qt * 16 + (lane & 15)) * 512 + h * 64;
  v16bf qa0 = load_a16(qrow, lane);
  v16bf qa1 = load_a16(qrow + 32, lane);
  const bf16_t* kbase = kb + (size_t)b * 1024 * 512 + h * 64;   // + key*512
  const bf16_t* vbase = vtb + ((size_t)b * 8 + h) * 64 * 1024;  // + d*1024 + key
  const unsigned int kl[2] = {(unsigned int)(uintptr_t)&kt_lds[0][0],
                              (unsigned int)(uintptr_t)&kt_lds[1][0]};
  const unsigned int vl[2] = {(unsigned int)(uintptr_t)&vt_lds[0][0],
                              (unsigned int)(uintptr_t)&vt_lds[1][0]};
  if (wid == 0) {
    tdm_load_2d(kbase, kl[0], 64, 32, 512);            // keys 0..31, 64 d-elems/row
    tdm_load_2d(vbase, vl[0], 32, 64, 1024);           // d 0..63, 32 keys/row
  }

  float mrow[8], lrow[8];
  v8f acc[4] = {v8f{}, v8f{}, v8f{}, v8f{}};
#pragma unroll
  for (int r = 0; r < 8; ++r) { mrow[r] = -__builtin_inff(); lrow[r] = 0.0f; }
  bf16_t* pst = plds[wid];

  for (int kt = 0; kt < 32; ++kt) {
    const int cur = kt & 1;
    if (wid == 0) {
      if (kt + 1 < 32) {
        const int key1 = (kt + 1) * 32;
        tdm_load_2d(kbase + (size_t)key1 * 512, kl[cur ^ 1], 64, 32, 512);
        tdm_load_2d(vbase + key1, vl[cur ^ 1], 32, 64, 1024);
        __builtin_amdgcn_s_wait_tensorcnt(2);          // older pair (cur) done
      } else {
        __builtin_amdgcn_s_wait_tensorcnt(0);
      }
    }
    __syncthreads();
    // ---- scores: S[16q x 32k] = Q(16x64) . K^T, K tile from LDS ----
    const bf16_t* krow0 = &kt_lds[cur][(lane & 15) * 64];
    const bf16_t* krow1 = krow0 + 16 * 64;
    v8f s0 = v8f{}, s1 = v8f{};
    {
      v16bf b00 = load_b16(krow0, lane);
      v16bf b10 = load_b16(krow1, lane);
      s0 = WMMA_BF16(qa0, b00, s0);
      s1 = WMMA_BF16(qa0, b10, s1);
      v16bf b01 = load_b16(krow0 + 32, lane);
      v16bf b11 = load_b16(krow1 + 32, lane);
      s0 = WMMA_BF16(qa1, b01, s0);
      s1 = WMMA_BF16(qa1, b11, s1);
    }
    // ---- online softmax over the 32-key tile ----
    float p0[8], p1[8], corr[8];
#pragma unroll
    for (int r = 0; r < 8; ++r) {
      p0[r] = s0[r] * scale;
      p1[r] = s1[r] * scale;
      float rmax = rowmax16(fmaxf(p0[r], p1[r]));
      float mnew = fmaxf(mrow[r], rmax);
      corr[r] = __expf(mrow[r] - mnew);
      mrow[r] = mnew;
      p0[r] = __expf(p0[r] - mnew);
      p1[r] = __expf(p1[r] - mnew);
      lrow[r] = lrow[r] * corr[r] + rowsum16(p0[r] + p1[r]);
    }
#pragma unroll
    for (int d = 0; d < 4; ++d)
#pragma unroll
      for (int r = 0; r < 8; ++r) acc[d][r] *= corr[r];
    // ---- relayout P (C layout) -> A layout via per-wave LDS ----
#pragma unroll
    for (int r = 0; r < 8; ++r) {
      const int m = r + ((lane >> 4) << 3);
      pst[m * 32 + (lane & 15)]      = (bf16_t)p0[r];
      pst[m * 32 + 16 + (lane & 15)] = (bf16_t)p1[r];
    }
    asm volatile("s_wait_dscnt 0" ::: "memory");       // stores visible before A reload
    v16bf pa = load_a16(pst + (lane & 15) * 32, lane);
    // ---- out += P(16x32) . V(32x64), V^T tile from LDS ----
#pragma unroll
    for (int d = 0; d < 4; ++d) {
      v16bf vb = load_b16(&vt_lds[cur][(d * 16 + (lane & 15)) * 32], lane);
      acc[d] = WMMA_BF16(pa, vb, acc[d]);
    }
    __syncthreads();                                   // all waves done with cur tiles
  }
  // ---- finalize: divide by denominator, store fp32 [b][s][h*64+d] ----
#pragma unroll
  for (int d = 0; d < 4; ++d) {
#pragma unroll
    for (int r = 0; r < 8; ++r) {
      const int q = qt * 16 + r + ((lane >> 4) << 3);
      const int col = h * 64 + d * 16 + (lane & 15);
      out[((size_t)b * 1024 + q) * 512 + col] = acc[d][r] / lrow[r];
    }
  }
}

extern "C" void kernel_launch(void* const* d_in, const int* in_sizes, int n_in,
                              void* d_out, int out_size, void* d_ws, size_t ws_size,
                              hipStream_t stream) {
  (void)in_sizes; (void)n_in; (void)out_size; (void)ws_size;
  const float* x   = (const float*)d_in[0];
  const float* Wcq = (const float*)d_in[1];  const float* bcq = (const float*)d_in[2];
  const float* Wck = (const float*)d_in[3];  const float* bck = (const float*)d_in[4];
  const float* Wcv = (const float*)d_in[5];  const float* bcv = (const float*)d_in[6];
  const float* Wpq = (const float*)d_in[7];  const float* bpq = (const float*)d_in[8];
  const float* Wpk = (const float*)d_in[9];  const float* bpk = (const float*)d_in[10];
  const float* Wpv = (const float*)d_in[11]; const float* bpv = (const float*)d_in[12];
  float* out = (float*)d_out;

  char* ws = (char*)d_ws;
  size_t off = 0;
  bf16_t* xb  = (bf16_t*)(ws + off); off += (size_t)8 * 1024 * 512 * 2;   // x in bf16
  bf16_t* wt  = (bf16_t*)(ws + off); off += (size_t)3 * 512 * 512 * 2;    // W^T bf16 (q,k,v)
  float*  bsm = (float*) (ws + off); off += (size_t)3 * 512 * 4 + 256;    // summed biases
  bf16_t* qb  = (bf16_t*)(ws + off); off += (size_t)8 * 1024 * 512 * 2;   // Q bf16 [b][s][d]
  bf16_t* kb  = (bf16_t*)(ws + off); off += (size_t)8 * 1024 * 512 * 2;   // K bf16 [b][s][d]
  bf16_t* vtb = (bf16_t*)(ws + off); off += (size_t)8 * 1024 * 512 * 2;   // V^T bf16 [b][h][d][s]

  da_prep_weights<<<3072, 256, 0, stream>>>(Wcq, Wpq, bcq, bpq,
                                            Wck, Wpk, bck, bpk,
                                            Wcv, Wpv, bcv, bpv, wt, bsm);
  da_convert_x<<<16384, 256, 0, stream>>>(x, xb);
  da_qkv_gemm<<<1536, 256, 0, stream>>>(xb, wt, bsm, qb, kb, vtb);
  da_attention<<<512, 256, 0, stream>>>(qb, kb, vtb, out);
}
